// Jurassic3Mamba_59004260712804
// MI455X (gfx1250) — matrix operand order repaired and checked by reading, not compiled, not run on hardware
//
#include <hip/hip_runtime.h>
#include <hip/hip_bf16.h>

// ---------------------------------------------------------------------------
// Mamba forward for MI455X (gfx1250, wave32).
// GEMMs: V_WMMA_F32_16X16X32_BF16, fp32 split into bf16 hi/lo (3 WMMAs per
// K=32 step) for ~fp32 accuracy at 3/8 the matrix-op cost of the native
// f32 16x16x4 path. Operands pre-packed into per-lane WMMA fragment order.
// Big GEMMs: 128x128 LDS-blocked, double-buffered, staged with CDNA5
// GLOBAL_LOAD_ASYNC_TO_LDS_B128 (ASYNCcnt) so DMA overlaps the WMMA burst.
// ---------------------------------------------------------------------------

typedef __attribute__((ext_vector_type(16))) __bf16 v16bf;
typedef __attribute__((ext_vector_type(8)))  float  v8f;

#define D_MODEL 2048
#define D_INNER 4096
#define D_STATE 16
#define D_CONV  4
#define DT_RANK 128
#define BATCH   2
#define SEQLEN  1024
#define NTOK    (BATCH * SEQLEN)          // 2048 tokens
#define XDBL_W  (DT_RANK + 2 * D_STATE)   // 160

static __device__ __forceinline__ float sigmoidf_(float x) {
  return 1.0f / (1.0f + __expf(-x));
}
static __device__ __forceinline__ float softplusf_(float x) {
  return (x > 20.0f) ? x : log1pf(__expf(x));
}

// ---- CDNA5 async global->LDS copy (ISA §15.18 opcode 98, ASYNCcnt) ----
// VDST carries the per-lane 32-bit LDS byte address (low 32 bits of the
// generic shared pointer per the LDS aperture rules).
static __device__ __forceinline__ void async_ld_b128(void* lds, const void* gsrc) {
  unsigned l = (unsigned)(size_t)lds;
  asm volatile("global_load_async_to_lds_b128 %0, %1, off"
               :: "v"(l), "v"(gsrc) : "memory");
}
static __device__ __forceinline__ void wait_async0() {
  asm volatile("s_wait_asynccnt 0x0" ::: "memory");
}

// ---------------------------------------------------------------------------
// Pack pass: convert f32 row-major [R x K] (leading dim lda) into a fragment
// buffer in per-lane WMMA order (ISA 7.12.2, wave32):
//   record = tr*Kt + tk ; per record: 32 lanes x 64B = [16 bf16 hi | 16 bf16 lo]
// IS_A layout: lane lo=M;  VGPR r -> K = 8*hi + (r<4 ? 2r : 16+2(r-4))
// B layout:    lane lo=N;  VGPR r -> K = 8*hi + (r<4 ? 2r : 16+2(r-4))
// ---------------------------------------------------------------------------
template <int IS_A>
__global__ __launch_bounds__(256) void pack_frag_kernel(
    const float* __restrict__ src, int lda,
    __bf16* __restrict__ dst, int Rt, int Kt) {
  const int gid = blockIdx.x * 256 + threadIdx.x;   // one thread = (record, lane)
  if (gid >= Rt * Kt * 32) return;
  const int lane = gid & 31;
  const int rec  = gid >> 5;
  const int tk = rec % Kt;
  const int tr = rec / Kt;
  const int lo = lane & 15, hi = lane >> 4;
  const float* p = src + (size_t)(tr * 16 + lo) * lda + tk * 32;

  v16bf fh, fl;
#pragma unroll
  for (int r = 0; r < 8; ++r) {
    const int kb = 8 * hi + ((r < 4) ? (2 * r) : (16 + 2 * (r - 4)));
    float2 v = *(const float2*)(p + kb);
    __bf16 h0 = (__bf16)v.x;            // native cvt, RNE
    __bf16 h1 = (__bf16)v.y;
    fh[2 * r]     = h0;
    fh[2 * r + 1] = h1;
    fl[2 * r]     = (__bf16)(v.x - (float)h0);
    fl[2 * r + 1] = (__bf16)(v.y - (float)h1);
  }
  __bf16* q = dst + (size_t)gid * 32;   // 64B per (record,lane)
  *(v16bf*)(q)      = fh;
  *(v16bf*)(q + 16) = fl;
}

#define WMMA3(ACC, AH, AL, BH, BL)                                                       \
  ACC = __builtin_amdgcn_wmma_f32_16x16x32_bf16(false, AH, false, BH, (short)0, ACC,     \
                                                false, false);                           \
  ACC = __builtin_amdgcn_wmma_f32_16x16x32_bf16(false, AH, false, BL, (short)0, ACC,     \
                                                false, false);                           \
  ACC = __builtin_amdgcn_wmma_f32_16x16x32_bf16(false, AL, false, BH, (short)0, ACC,     \
                                                false, false);

// ---------------------------------------------------------------------------
// LDS-blocked GEMM (M,N multiples of 128): block = 8 waves (4M x 2N) computing
// a 128x128 tile. Per K=32 step: async-DMA 16 records (32KB) global->LDS,
// double-buffered; waves read fragments from LDS (ds_load_b128) and issue
// 24 WMMAs each. The async copy for step tk+1 is in flight during step tk's
// WMMA burst; one s_wait_asynccnt + barrier per step.
// EPI==1: C = softplus(C + bias[n]).
// ---------------------------------------------------------------------------
template <int EPI>
__global__ __launch_bounds__(256) void wmma_gemm_lds_kernel(
    const __bf16* __restrict__ Af,   // [Mt x Kt] records
    const __bf16* __restrict__ Bf,   // [Nt x Kt] records
    float* __restrict__ C, int ldc,
    const float* __restrict__ bias,
    int MtB, int NtB, int Kt) {      // MtB = M/128, NtB = N/128
  __shared__ float4 sA[2][1024];     // 2 x 8 records x 2KB = 32KB
  __shared__ float4 sB[2][1024];     // 32KB
  const int bidm = blockIdx.x % MtB;
  const int bidn = blockIdx.x / MtB;
  const int tid  = threadIdx.x;
  const int wave = tid >> 5;
  const int lane = tid & 31;
  const int wm = wave >> 1;          // 0..3 -> m-tiles 2wm, 2wm+1
  const int wn = wave & 1;           // 0..1 -> n-tiles 4wn .. 4wn+3
  const size_t tstride = (size_t)Kt * 1024;          // elems per tile row

  // staging map: thread DMAs 4 A chunks + 4 B chunks of 16B each
  const int off = tid & 127;         // chunk-in-record (x16B)
  const int r0  = tid >> 7;          // 0 or 1; records r0, r0+2, r0+4, r0+6
  const __bf16* gA = Af + (size_t)(bidm * 8) * tstride + (size_t)off * 8;
  const __bf16* gB = Bf + (size_t)(bidn * 8) * tstride + (size_t)off * 8;

  v8f acc[2][4];
#pragma unroll
  for (int mi = 0; mi < 2; ++mi)
#pragma unroll
    for (int j = 0; j < 4; ++j)
      acc[mi][j] = (v8f){0.f, 0.f, 0.f, 0.f, 0.f, 0.f, 0.f, 0.f};

  // ---- prologue: async-stage tk = 0 into buffer 0 ----
#pragma unroll
  for (int q = 0; q < 4; ++q) {
    const int rec = r0 + 2 * q;
    async_ld_b128(&sA[0][rec * 128 + off], gA + (size_t)rec * tstride);
    async_ld_b128(&sB[0][rec * 128 + off], gB + (size_t)rec * tstride);
  }
  wait_async0();
  __syncthreads();

  int s = 0;
  for (int tk = 0; tk < Kt; ++tk) {
    // ---- launch next stage's async DMA into the other buffer ----
    if (tk + 1 < Kt) {
      const size_t koff = (size_t)(tk + 1) * 1024;
#pragma unroll
      for (int q = 0; q < 4; ++q) {
        const int rec = r0 + 2 * q;
        async_ld_b128(&sA[s ^ 1][rec * 128 + off], gA + (size_t)rec * tstride + koff);
        async_ld_b128(&sB[s ^ 1][rec * 128 + off], gB + (size_t)rec * tstride + koff);
      }
    }
    // ---- compute from LDS buffer s ----
    const __bf16* la0 = (const __bf16*)sA[s] + (2 * wm) * 1024 + lane * 32;
    const __bf16* la1 = la0 + 1024;
    const __bf16* lb0 = (const __bf16*)sB[s] + (4 * wn) * 1024 + lane * 32;
    const __bf16* lb1 = lb0 + 1024;
    const __bf16* lb2 = lb1 + 1024;
    const __bf16* lb3 = lb2 + 1024;
    const v16bf a0h = *(const v16bf*)(la0);
    const v16bf a0l = *(const v16bf*)(la0 + 16);
    const v16bf a1h = *(const v16bf*)(la1);
    const v16bf a1l = *(const v16bf*)(la1 + 16);
    const v16bf b0h = *(const v16bf*)(lb0);
    const v16bf b0l = *(const v16bf*)(lb0 + 16);
    const v16bf b1h = *(const v16bf*)(lb1);
    const v16bf b1l = *(const v16bf*)(lb1 + 16);
    const v16bf b2h = *(const v16bf*)(lb2);
    const v16bf b2l = *(const v16bf*)(lb2 + 16);
    const v16bf b3h = *(const v16bf*)(lb3);
    const v16bf b3l = *(const v16bf*)(lb3 + 16);

    WMMA3(acc[0][0], a0h, a0l, b0h, b0l)
    WMMA3(acc[1][0], a1h, a1l, b0h, b0l)
    WMMA3(acc[0][1], a0h, a0l, b1h, b1l)
    WMMA3(acc[1][1], a1h, a1l, b1h, b1l)
    WMMA3(acc[0][2], a0h, a0l, b2h, b2l)
    WMMA3(acc[1][2], a1h, a1l, b2h, b2l)
    WMMA3(acc[0][3], a0h, a0l, b3h, b3l)
    WMMA3(acc[1][3], a1h, a1l, b3h, b3l)

    // ---- ensure our DMA landed, then flip buffers block-wide ----
    if (tk + 1 < Kt) wait_async0();
    __syncthreads();
    s ^= 1;
  }

  // C/D layout: lane lo = N col; VGPR v -> M = v + 8*hi
  const int lo = lane & 15, hi = lane >> 4;
#pragma unroll
  for (int j = 0; j < 4; ++j) {
    const int col = ((bidn * 8 + 4 * wn + j) << 4) + lo;
    const float bv = (EPI == 1) ? bias[col] : 0.0f;
#pragma unroll
    for (int mi = 0; mi < 2; ++mi) {
      const int rbase = ((bidm * 8 + 2 * wm + mi) << 4) + (hi << 3);
#pragma unroll
      for (int v = 0; v < 8; ++v) {
        float val = acc[mi][j][v];
        if (EPI == 1) val = softplusf_(val + bv);
        C[(size_t)(rbase + v) * ldc + col] = val;
      }
    }
  }
}

// ---------------------------------------------------------------------------
// Register-direct GEMM (for ragged N, e.g. x_proj N=160): wave tile 32x64.
// ---------------------------------------------------------------------------
__global__ __launch_bounds__(128) void wmma_gemm_reg_kernel(
    const __bf16* __restrict__ Af, const __bf16* __restrict__ Bf,
    float* __restrict__ C, int ldc,
    int Mt2, int Nt4, int Kt) {
  const int wid = blockIdx.x * 4 + (threadIdx.x >> 5);
  if (wid >= Mt2 * Nt4) return;             // wave-uniform exit
  const int tm0 = (wid % Mt2) << 1;
  const int tn0 = (wid / Mt2) << 2;
  const int lane = threadIdx.x & 31;

  v8f acc[2][4];
#pragma unroll
  for (int mi = 0; mi < 2; ++mi)
#pragma unroll
    for (int j = 0; j < 4; ++j)
      acc[mi][j] = (v8f){0.f, 0.f, 0.f, 0.f, 0.f, 0.f, 0.f, 0.f};

  const size_t tstride = (size_t)Kt * 1024;
  const __bf16* pa0 = Af + (size_t)tm0 * tstride + (size_t)lane * 32;
  const __bf16* pa1 = pa0 + tstride;
  const __bf16* pb0 = Bf + (size_t)tn0 * tstride + (size_t)lane * 32;
  const __bf16* pb1 = pb0 + tstride;
  const __bf16* pb2 = pb1 + tstride;
  const __bf16* pb3 = pb2 + tstride;

  for (int tk = 0; tk < Kt; ++tk) {
    const v16bf a0h = *(const v16bf*)(pa0);
    const v16bf a0l = *(const v16bf*)(pa0 + 16);
    const v16bf a1h = *(const v16bf*)(pa1);
    const v16bf a1l = *(const v16bf*)(pa1 + 16);
    const v16bf b0h = *(const v16bf*)(pb0);
    const v16bf b0l = *(const v16bf*)(pb0 + 16);
    const v16bf b1h = *(const v16bf*)(pb1);
    const v16bf b1l = *(const v16bf*)(pb1 + 16);
    const v16bf b2h = *(const v16bf*)(pb2);
    const v16bf b2l = *(const v16bf*)(pb2 + 16);
    const v16bf b3h = *(const v16bf*)(pb3);
    const v16bf b3l = *(const v16bf*)(pb3 + 16);
    pa0 += 1024; pa1 += 1024;
    pb0 += 1024; pb1 += 1024; pb2 += 1024; pb3 += 1024;

    WMMA3(acc[0][0], a0h, a0l, b0h, b0l)
    WMMA3(acc[1][0], a1h, a1l, b0h, b0l)
    WMMA3(acc[0][1], a0h, a0l, b1h, b1l)
    WMMA3(acc[1][1], a1h, a1l, b1h, b1l)
    WMMA3(acc[0][2], a0h, a0l, b2h, b2l)
    WMMA3(acc[1][2], a1h, a1l, b2h, b2l)
    WMMA3(acc[0][3], a0h, a0l, b3h, b3l)
    WMMA3(acc[1][3], a1h, a1l, b3h, b3l)
  }

  const int lo = lane & 15, hi = lane >> 4;
#pragma unroll
  for (int j = 0; j < 4; ++j) {
    const int col = ((tn0 + j) << 4) + lo;
#pragma unroll
    for (int mi = 0; mi < 2; ++mi) {
      const int rbase = ((tm0 + mi) << 4) + (hi << 3);
#pragma unroll
      for (int v = 0; v < 8; ++v)
        C[(size_t)(rbase + v) * ldc + col] = acc[mi][j][v];
    }
  }
}

// Tail kernel: one 16-col N tile per wave (32x16), for Nt % 4 remainder.
__global__ __launch_bounds__(128) void wmma_gemm_tail_kernel(
    const __bf16* __restrict__ Af, const __bf16* __restrict__ Bf,
    float* __restrict__ C, int ldc,
    int Mt2, int Kt, int tnStart, int nTail) {
  const int wid = blockIdx.x * 4 + (threadIdx.x >> 5);
  if (wid >= Mt2 * nTail) return;
  const int tm0 = (wid % Mt2) << 1;
  const int tn  = tnStart + (wid / Mt2);
  const int lane = threadIdx.x & 31;

  v8f acc0 = (v8f){0.f, 0.f, 0.f, 0.f, 0.f, 0.f, 0.f, 0.f};
  v8f acc1 = acc0;
  const size_t tstride = (size_t)Kt * 1024;
  const __bf16* pa0 = Af + (size_t)tm0 * tstride + (size_t)lane * 32;
  const __bf16* pa1 = pa0 + tstride;
  const __bf16* pb  = Bf + (size_t)tn * tstride + (size_t)lane * 32;

  for (int tk = 0; tk < Kt; ++tk) {
    const v16bf a0h = *(const v16bf*)(pa0);
    const v16bf a0l = *(const v16bf*)(pa0 + 16);
    const v16bf a1h = *(const v16bf*)(pa1);
    const v16bf a1l = *(const v16bf*)(pa1 + 16);
    const v16bf bh  = *(const v16bf*)(pb);
    const v16bf bl  = *(const v16bf*)(pb + 16);
    pa0 += 1024; pa1 += 1024; pb += 1024;
    WMMA3(acc0, a0h, a0l, bh, bl)
    WMMA3(acc1, a1h, a1l, bh, bl)
  }

  const int lo = lane & 15, hi = lane >> 4;
  const int col = (tn << 4) + lo;
#pragma unroll
  for (int v = 0; v < 8; ++v) {
    C[(size_t)((tm0 << 4) + (hi << 3) + v) * ldc + col] = acc0[v];
    C[(size_t)(((tm0 + 1) << 4) + (hi << 3) + v) * ldc + col] = acc1[v];
  }
}

// ---------- causal depthwise conv1d (K=4) + SiLU ----------
__global__ __launch_bounds__(256) void conv_silu_kernel(
    const float* __restrict__ xz,        // [NTOK, 2*D_INNER]; x = cols [0,4096)
    const float* __restrict__ conv_w,    // [D_INNER, 4]
    const float* __restrict__ conv_b,    // [D_INNER]
    float* __restrict__ x_act) {         // [NTOK, D_INNER]
  const int idx = blockIdx.x * blockDim.x + threadIdx.x;
  if (idx >= NTOK * D_INNER) return;
  const int d = idx & (D_INNER - 1);
  const int t = idx >> 12;               // D_INNER = 2^12
  const int b = t >> 10;                 // SEQLEN = 2^10
  const int l = t & (SEQLEN - 1);
  float acc = conv_b[d];
#pragma unroll
  for (int k = 0; k < D_CONV; ++k) {
    const int ls = l - (D_CONV - 1) + k;
    if (ls >= 0)
      acc += xz[(size_t)(b * SEQLEN + ls) * (2 * D_INNER) + d] * conv_w[d * D_CONV + k];
  }
  x_act[idx] = acc * sigmoidf_(acc);
}

// ---------------------------------------------------------------------------
// Selective scan: lane = one (b,d) channel; h[16] in VGPRs; sequential in L.
// B_t/C_t read per-lane as float4 (same cacheline -> HW broadcast).
// Software-pipelined: step l+1 loads issued before step l compute.
// Fused: y = (scan_y + D*x) * silu(z)
// ---------------------------------------------------------------------------
__global__ __launch_bounds__(256) void scan_kernel(
    const float* __restrict__ xz,        // z = cols [4096,8192)
    const float* __restrict__ x_act,
    const float* __restrict__ x_dbl,     // [NTOK, 160]: dt|B|C
    const float* __restrict__ dtb,       // [NTOK, D_INNER] (post softplus)
    const float* __restrict__ A_log,     // [D_INNER, 16]
    const float* __restrict__ Dvec,      // [D_INNER]
    float* __restrict__ ybuf) {          // [NTOK, D_INNER]
  const int ch = blockIdx.x * 256 + threadIdx.x;       // global channel
  const int b = ch >> 12;                              // 4096 channels per batch
  const int d = ch & (D_INNER - 1);

  float Arow[D_STATE], h[D_STATE];
#pragma unroll
  for (int n = 0; n < D_STATE; ++n) {
    Arow[n] = -__expf(A_log[d * D_STATE + n]);
    h[n] = 0.0f;
  }
  const float Dd = Dvec[d];

  const int t0 = b * SEQLEN;
  float x   = x_act[(size_t)t0 * D_INNER + d];
  float dtv = dtb[(size_t)t0 * D_INNER + d];
  float zv  = xz[(size_t)t0 * (2 * D_INNER) + D_INNER + d];
  const float4* pbc = (const float4*)(x_dbl + (size_t)t0 * XDBL_W + DT_RANK);
  float4 B01 = pbc[0], B23 = pbc[1], B45 = pbc[2], B67 = pbc[3];
  float4 C01 = pbc[4], C23 = pbc[5], C45 = pbc[6], C67 = pbc[7];

  for (int l = 0; l < SEQLEN; ++l) {
    const int t = t0 + l;
    float nx = 0.f, ndt = 0.f, nz = 0.f;
    float4 nB01 = {}, nB23 = {}, nB45 = {}, nB67 = {};
    float4 nC01 = {}, nC23 = {}, nC45 = {}, nC67 = {};
    if (l + 1 < SEQLEN) {
      const int tn = t + 1;
      nx  = x_act[(size_t)tn * D_INNER + d];
      ndt = dtb[(size_t)tn * D_INNER + d];
      nz  = xz[(size_t)tn * (2 * D_INNER) + D_INNER + d];
      const float4* pn = (const float4*)(x_dbl + (size_t)tn * XDBL_W + DT_RANK);
      nB01 = pn[0]; nB23 = pn[1]; nB45 = pn[2]; nB67 = pn[3];
      nC01 = pn[4]; nC23 = pn[5]; nC45 = pn[6]; nC67 = pn[7];
    }
    const float Bv[D_STATE] = {B01.x, B01.y, B01.z, B01.w, B23.x, B23.y, B23.z, B23.w,
                               B45.x, B45.y, B45.z, B45.w, B67.x, B67.y, B67.z, B67.w};
    const float Cv[D_STATE] = {C01.x, C01.y, C01.z, C01.w, C23.x, C23.y, C23.z, C23.w,
                               C45.x, C45.y, C45.z, C45.w, C67.x, C67.y, C67.z, C67.w};
    const float dx = dtv * x;
    float y = 0.0f;
#pragma unroll
    for (int n = 0; n < D_STATE; ++n) {
      const float dA = __expf(dtv * Arow[n]);
      h[n] = h[n] * dA + dx * Bv[n];
      y = fmaf(h[n], Cv[n], y);
    }
    y = fmaf(Dd, x, y);
    y *= zv * sigmoidf_(zv);
    ybuf[(size_t)t * D_INNER + d] = y;
    x = nx; dtv = ndt; zv = nz;
    B01 = nB01; B23 = nB23; B45 = nB45; B67 = nB67;
    C01 = nC01; C23 = nC23; C45 = nC45; C67 = nC67;
  }
}

// ---------------------------------------------------------------------------
extern "C" void kernel_launch(void* const* d_in, const int* in_sizes, int n_in,
                              void* d_out, int out_size, void* d_ws, size_t ws_size,
                              hipStream_t stream) {
  (void)in_sizes; (void)n_in; (void)out_size; (void)ws_size;
  const float* hs        = (const float*)d_in[0];  // [2,1024,2048]
  const float* in_proj_w = (const float*)d_in[1];  // [8192,2048]
  const float* conv_w    = (const float*)d_in[2];  // [4096,4]
  const float* conv_b    = (const float*)d_in[3];  // [4096]
  const float* x_proj_w  = (const float*)d_in[4];  // [160,4096]
  const float* dt_proj_w = (const float*)d_in[5];  // [4096,128]
  const float* dt_proj_b = (const float*)d_in[6];  // [4096]
  const float* A_log     = (const float*)d_in[7];  // [4096,16]
  const float* Dvec      = (const float*)d_in[8];  // [4096]
  const float* out_proj_w= (const float*)d_in[9];  // [2048,4096]
  float* out = (float*)d_out;                      // [2,1024,2048]

  char* wsb = (char*)d_ws;
  float* xz    = (float*)wsb;                       wsb += (size_t)NTOK * 2 * D_INNER * 4;
  float* x_act = (float*)wsb;                       wsb += (size_t)NTOK * D_INNER * 4;
  float* x_dbl = (float*)wsb;                       wsb += (size_t)NTOK * XDBL_W * 4;
  float* dtb   = (float*)wsb;                       wsb += (size_t)NTOK * D_INNER * 4;
  float* ybuf  = (float*)wsb;                       wsb += (size_t)NTOK * D_INNER * 4;
  __bf16* fragA = (__bf16*)wsb;                     wsb += (size_t)NTOK * D_INNER * 4;
  __bf16* fragB = (__bf16*)wsb;

  auto pack = [&](const float* src, int lda, int R, int K, __bf16* dst, bool isA) {
    const int Rt = R / 16, Kt = K / 32;
    const int blocks = (Rt * Kt * 32 + 255) / 256;
    if (isA) pack_frag_kernel<1><<<blocks, 256, 0, stream>>>(src, lda, dst, Rt, Kt);
    else     pack_frag_kernel<0><<<blocks, 256, 0, stream>>>(src, lda, dst, Rt, Kt);
  };
  auto gemm = [&](int M, int N, int K, const __bf16* Af, const __bf16* Bf,
                  float* C, int ldc, const float* bias, int epi) {
    const int Kt = K / 32;
    if ((M % 128 == 0) && (N % 128 == 0)) {
      const int MtB = M / 128, NtB = N / 128;
      if (epi)
        wmma_gemm_lds_kernel<1><<<MtB * NtB, 256, 0, stream>>>(Af, Bf, C, ldc, bias, MtB, NtB, Kt);
      else
        wmma_gemm_lds_kernel<0><<<MtB * NtB, 256, 0, stream>>>(Af, Bf, C, ldc, bias, MtB, NtB, Kt);
    } else {
      const int Mt2 = M / 32, Nt = N / 16;
      const int Nt4 = Nt / 4, nTail = Nt % 4;
      if (Nt4 > 0) {
        const int blocks = (Mt2 * Nt4 + 3) / 4;
        wmma_gemm_reg_kernel<<<blocks, 128, 0, stream>>>(Af, Bf, C, ldc, Mt2, Nt4, Kt);
      }
      if (nTail > 0) {
        const int blocks = (Mt2 * nTail + 3) / 4;
        wmma_gemm_tail_kernel<<<blocks, 128, 0, stream>>>(Af, Bf, C, ldc, Mt2, Kt, Nt4 * 4, nTail);
      }
    }
  };

  // 1) in_proj: xz = hs @ in_proj_w^T   (M=2048, N=8192, K=2048)
  pack(hs, D_MODEL, NTOK, D_MODEL, fragA, true);
  pack(in_proj_w, D_MODEL, 2 * D_INNER, D_MODEL, fragB, false);
  gemm(NTOK, 2 * D_INNER, D_MODEL, fragA, fragB, xz, 2 * D_INNER, nullptr, 0);

  // 2) causal conv1d + SiLU -> x_act
  conv_silu_kernel<<<(NTOK * D_INNER + 255) / 256, 256, 0, stream>>>(
      xz, conv_w, conv_b, x_act);

  // 3) x_proj: x_dbl = x_act @ x_proj_w^T   (M=2048, N=160, K=4096) - reg path
  pack(x_act, D_INNER, NTOK, D_INNER, fragA, true);
  pack(x_proj_w, D_INNER, XDBL_W, D_INNER, fragB, false);
  gemm(NTOK, XDBL_W, D_INNER, fragA, fragB, x_dbl, XDBL_W, nullptr, 0);

  // 4) dt_proj + bias + softplus  (M=2048, N=4096, K=128; A = x_dbl[:, :128], lda=160)
  pack(x_dbl, XDBL_W, NTOK, DT_RANK, fragA, true);
  pack(dt_proj_w, DT_RANK, D_INNER, DT_RANK, fragB, false);
  gemm(NTOK, D_INNER, DT_RANK, fragA, fragB, dtb, D_INNER, dt_proj_b, 1);

  // 5) selective scan + D-skip + z-gating -> ybuf  (8192 channels, 32 blocks)
  scan_kernel<<<(BATCH * D_INNER) / 256, 256, 0, stream>>>(
      xz, x_act, x_dbl, dtb, A_log, Dvec, ybuf);

  // 6) out_proj: out = ybuf @ out_proj_w^T   (M=2048, N=2048, K=4096)
  pack(ybuf, D_INNER, NTOK, D_INNER, fragA, true);
  pack(out_proj_w, D_INNER, D_MODEL, D_INNER, fragB, false);
  gemm(NTOK, D_MODEL, D_INNER, fragA, fragB, out, D_MODEL, nullptr, 0);
}
